// Processor_1589137899997
// MI455X (gfx1250) — matrix operand order repaired and checked by reading, not compiled, not run on hardware
//
#include <hip/hip_runtime.h>

#define NNODES 50000
#define NEDGES 640000
#define DF 128          // D == ED == H == 128

typedef __attribute__((ext_vector_type(16))) __bf16 v16bf;
typedef __attribute__((ext_vector_type(8)))  __bf16 v8bf;
typedef __attribute__((ext_vector_type(8)))  float  v8f;

// ---- helpers -------------------------------------------------------------

union U16bf { v16bf v; v8bf h[2]; uint4 q[2]; };

// B fragment: two 8-bf16 chunks 16 elements apart (two ds_load_b128)
static __device__ __forceinline__ v16bf lds_b(const __bf16* p) {
  U16bf u;
  u.q[0] = *reinterpret_cast<const uint4*>(p);
  u.q[1] = *reinterpret_cast<const uint4*>(p + 16);
  return u.v;
}
// A fragment from LDS: two 8-bf16 chunks 16 elements apart
static __device__ __forceinline__ v16bf lds_a(const __bf16* p) {
  U16bf u;
  u.q[0] = *reinterpret_cast<const uint4*>(p);
  u.q[1] = *reinterpret_cast<const uint4*>(p + 16);
  return u.v;
}

// 8 f32 -> 8 bf16 via hardware packed converts (v_cvt_pk_bf16_f32)
static __device__ __forceinline__ v8bf cvt8v(const float* p) {
  v8f f = *reinterpret_cast<const v8f*>(p);
  return __builtin_convertvector(f, v8bf);
}
static __device__ __forceinline__ v8bf cvt8vs(const float* p, float s) {
  v8f f = *reinterpret_cast<const v8f*>(p);
  f = f * s;
  return __builtin_convertvector(f, v8bf);
}

static __device__ __forceinline__ v8f vzero8() {
  v8f a;
#pragma unroll
  for (int i = 0; i < 8; ++i) a[i] = 0.f;
  return a;
}

static __device__ __forceinline__ v8f wmma_bf16(v16bf a, v16bf b, v8f c) {
  // D(16x16 f32) = A(16x32 bf16) x B(32x16 bf16) + C
  return __builtin_amdgcn_wmma_f32_16x16x32_bf16(false, a, false, b, (short)0, c,
                                                 false, false);
}

// ---- kernel 0: zero the aggregation buffers ------------------------------

__global__ void gnn_zero_kernel(float* __restrict__ agg, float* __restrict__ cnt) {
  size_t i = (size_t)blockIdx.x * 256 + threadIdx.x;
  if (i < (size_t)NNODES * DF) agg[i] = 0.f;
  if (i < NNODES)              cnt[i] = 0.f;
}

// ---- kernel 1: edge MLP + scatter-add ------------------------------------
// LDS: We1^T bf16 [128][384+8], We2^T bf16 [128][128+8], per-wave h tile.

__global__ void __launch_bounds__(256) gnn_edge_kernel(
    const float* __restrict__ x,   const int* __restrict__ ei,
    const float* __restrict__ ea,
    const float* __restrict__ We1, const float* __restrict__ be1,
    const float* __restrict__ We2, const float* __restrict__ be2,
    float* __restrict__ eout, float* agg, float* cnt)
{
  __shared__ __bf16 sW1[128 * 392];     // stride 392 bf16 (conflict-free)
  __shared__ __bf16 sW2[128 * 136];
  __shared__ __bf16 sH [8 * 16 * 136];  // per-wave 16x128 hidden tile

  for (int idx = threadIdx.x; idx < 384 * 128; idx += 256) {
    int k = idx >> 7, n = idx & 127;
    sW1[n * 392 + k] = (__bf16)We1[idx];  // transpose: [N][K]
  }
  for (int idx = threadIdx.x; idx < 128 * 128; idx += 256) {
    int k = idx >> 7, n = idx & 127;
    sW2[n * 136 + k] = (__bf16)We2[idx];
  }
  __syncthreads();

  const int wave = threadIdx.x >> 5;
  const int lane = threadIdx.x & 31;
  const int m    = lane & 15;           // tile row (M) / column within ct (N)
  const int half = lane >> 4;           // K-half selector for A/B fragments
  __bf16* hbase = &sH[wave * 16 * 136];

  const int nMacro = (NEDGES / 16) / 8; // 5000 macro tiles of 8 waves x 16 edges
  for (int mb = blockIdx.x; mb < nMacro; mb += gridDim.x) {
    const int tile  = mb * 8 + wave;
    const int ebase = tile * 16;
    const int e     = ebase + m;
    const int rown  = ei[e];            // aggregation target
    const int coln  = ei[NEDGES + e];
    const float* seg0 = x  + (size_t)rown * DF;
    const float* seg1 = x  + (size_t)coln * DF;
    const float* seg2 = ea + (size_t)e    * DF;

    // prefetch next macro-tile's edge_attr row + indices (global_prefetch_b8)
    {
      const size_t enext = (size_t)e + (size_t)gridDim.x * 128;
      if (enext < NEDGES) {
        __builtin_prefetch(ea + enext * DF, 0, 1);
        __builtin_prefetch(ei + enext, 0, 1);
        __builtin_prefetch(ei + NEDGES + enext, 0, 1);
      }
    }

    // ---- layer 1: [16,384] @ [384,128] ----
    v8f acc[8];
#pragma unroll
    for (int ct = 0; ct < 8; ++ct) acc[ct] = vzero8();

#pragma unroll
    for (int s = 0; s < 12; ++s) {      // K = 12 * 32
      const int K32 = s * 32;
      const float* sp = (s < 4 ? seg0 : (s < 8 ? seg1 : seg2))
                        + (K32 & 127) + half * 8;
      U16bf A;
      A.h[0] = cvt8v(sp);               // K = half*8 + 0..7
      A.h[1] = cvt8v(sp + 16);          // K = 16 + half*8 + 0..7
      // load all 8 B fragments, then issue 8 back-to-back WMMAs
      v16bf Bf[8];
#pragma unroll
      for (int ct = 0; ct < 8; ++ct)
        Bf[ct] = lds_b(&sW1[(ct * 16 + m) * 392 + K32 + half * 8]);
#pragma unroll
      for (int ct = 0; ct < 8; ++ct)
        acc[ct] = wmma_bf16(A.v, Bf[ct], acc[ct]);
    }

    // bias + ReLU -> bf16 h tile in LDS (re-layout D -> A for layer 2)
#pragma unroll
    for (int ct = 0; ct < 8; ++ct) {
      const int n  = ct * 16 + m;
      const float b1 = be1[n];
#pragma unroll
      for (int rr = 0; rr < 8; ++rr) {
        float v = acc[ct][rr] + b1;
        v = v > 0.f ? v : 0.f;
        hbase[(rr + 8 * half) * 136 + n] = (__bf16)v;
      }
    }
    asm volatile("s_wait_dscnt 0" ::: "memory");  // wave-local store->load

    // ---- layer 2: [16,128] @ [128,128] ----
    v8f acc2[8];
#pragma unroll
    for (int ct = 0; ct < 8; ++ct) acc2[ct] = vzero8();
#pragma unroll
    for (int s = 0; s < 4; ++s) {
      const int K32 = s * 32;
      v16bf A2 = lds_a(&hbase[m * 136 + K32 + half * 8]);
      v16bf Bf[8];
#pragma unroll
      for (int ct = 0; ct < 8; ++ct)
        Bf[ct] = lds_b(&sW2[(ct * 16 + m) * 136 + K32 + half * 8]);
#pragma unroll
      for (int ct = 0; ct < 8; ++ct)
        acc2[ct] = wmma_bf16(A2, Bf[ct], acc2[ct]);
    }

    // scatter targets per D-row (depends only on rr) -- hoisted out of ct loop
    int tgt[8];
#pragma unroll
    for (int rr = 0; rr < 8; ++rr) tgt[rr] = __shfl(rown, rr + 8 * half, 32);

    // epilogue: bias, write edge_out, scatter-add into agg
#pragma unroll
    for (int ct = 0; ct < 8; ++ct) {
      const int n  = ct * 16 + m;
      const float b2 = be2[n];
#pragma unroll
      for (int rr = 0; rr < 8; ++rr) {
        const int mrow = rr + 8 * half;
        const int ee   = ebase + mrow;
        float v = acc2[ct][rr] + b2;
        eout[(size_t)ee * DF + n] = v;
        atomicAdd(&agg[(size_t)tgt[rr] * DF + n], v);
      }
    }
    if (half == 0) atomicAdd(&cnt[rown], 1.0f);   // one lane per edge
  }
}

// ---- kernel 2: node MLP (reads agg in place, overwrites with x_out) ------

__global__ void __launch_bounds__(256) gnn_node_kernel(
    const float* __restrict__ x,
    const float* __restrict__ Wn1, const float* __restrict__ bn1,
    const float* __restrict__ Wn2, const float* __restrict__ bn2,
    const float* agg, const float* __restrict__ cnt, float* xout)
{
  __shared__ __bf16 sW1[128 * 264];     // K=256 -> stride 264
  __shared__ __bf16 sW2[128 * 136];
  __shared__ __bf16 sH [8 * 16 * 136];

  for (int idx = threadIdx.x; idx < 256 * 128; idx += 256) {
    int k = idx >> 7, n = idx & 127;
    sW1[n * 264 + k] = (__bf16)Wn1[idx];
  }
  for (int idx = threadIdx.x; idx < 128 * 128; idx += 256) {
    int k = idx >> 7, n = idx & 127;
    sW2[n * 136 + k] = (__bf16)Wn2[idx];
  }
  __syncthreads();

  const int wave = threadIdx.x >> 5;
  const int lane = threadIdx.x & 31;
  const int m    = lane & 15;
  const int half = lane >> 4;
  __bf16* hbase = &sH[wave * 16 * 136];

  const int tile = blockIdx.x * 8 + wave;       // 3125 tiles total
  if (tile >= NNODES / 16) return;              // wave-uniform exit
  const int nbase = tile * 16;
  const int node  = nbase + m;
  const float cinv = 1.0f / fmaxf(cnt[node], 1.0f);
  const float* seg0 = x   + (size_t)node * DF;
  const float* seg1 = agg + (size_t)node * DF;  // scatter-sum -> mean via cinv

  v8f acc[8];
#pragma unroll
  for (int ct = 0; ct < 8; ++ct) acc[ct] = vzero8();

#pragma unroll
  for (int s = 0; s < 8; ++s) {                 // K = 8 * 32 = 256
    const int K32 = s * 32;
    U16bf A;
    if (s < 4) {
      const float* sp = seg0 + K32 + half * 8;
      A.h[0] = cvt8v(sp);
      A.h[1] = cvt8v(sp + 16);
    } else {
      const float* sp = seg1 + (K32 - 128) + half * 8;
      A.h[0] = cvt8vs(sp, cinv);
      A.h[1] = cvt8vs(sp + 16, cinv);
    }
    v16bf Bf[8];
#pragma unroll
    for (int ct = 0; ct < 8; ++ct)
      Bf[ct] = lds_b(&sW1[(ct * 16 + m) * 264 + K32 + half * 8]);
#pragma unroll
    for (int ct = 0; ct < 8; ++ct)
      acc[ct] = wmma_bf16(A.v, Bf[ct], acc[ct]);
  }

#pragma unroll
  for (int ct = 0; ct < 8; ++ct) {
    const int n = ct * 16 + m;
    const float b1 = bn1[n];
#pragma unroll
    for (int rr = 0; rr < 8; ++rr) {
      float v = acc[ct][rr] + b1;
      v = v > 0.f ? v : 0.f;
      hbase[(rr + 8 * half) * 136 + n] = (__bf16)v;
    }
  }
  asm volatile("s_wait_dscnt 0" ::: "memory");

  v8f acc2[8];
#pragma unroll
  for (int ct = 0; ct < 8; ++ct) acc2[ct] = vzero8();
#pragma unroll
  for (int s = 0; s < 4; ++s) {
    const int K32 = s * 32;
    v16bf A2 = lds_a(&hbase[m * 136 + K32 + half * 8]);
    v16bf Bf[8];
#pragma unroll
    for (int ct = 0; ct < 8; ++ct)
      Bf[ct] = lds_b(&sW2[(ct * 16 + m) * 136 + K32 + half * 8]);
#pragma unroll
    for (int ct = 0; ct < 8; ++ct)
      acc2[ct] = wmma_bf16(A2, Bf[ct], acc2[ct]);
  }

  asm volatile("s_wait_loadcnt 0" ::: "memory"); // agg reads done before overwrite
#pragma unroll
  for (int ct = 0; ct < 8; ++ct) {
    const int n = ct * 16 + m;
    const float b2 = bn2[n];
#pragma unroll
    for (int rr = 0; rr < 8; ++rr) {
      const int mrow = rr + 8 * half;
      xout[(size_t)(nbase + mrow) * DF + n] = acc2[ct][rr] + b2;
    }
  }
}

// ---- launch --------------------------------------------------------------

extern "C" void kernel_launch(void* const* d_in, const int* in_sizes, int n_in,
                              void* d_out, int out_size, void* d_ws, size_t ws_size,
                              hipStream_t stream) {
  (void)in_sizes; (void)n_in; (void)out_size; (void)ws_size;
  const float* x   = (const float*)d_in[0];
  const int*   ei  = (const int*)  d_in[1];
  const float* ea  = (const float*)d_in[2];
  const float* We1 = (const float*)d_in[3];
  const float* be1 = (const float*)d_in[4];
  const float* We2 = (const float*)d_in[5];
  const float* be2 = (const float*)d_in[6];
  const float* Wn1 = (const float*)d_in[7];
  const float* bn1 = (const float*)d_in[8];
  const float* Wn2 = (const float*)d_in[9];
  const float* bn2 = (const float*)d_in[10];

  float* xout = (float*)d_out;                   // [N,128]; doubles as agg buffer
  float* eout = xout + (size_t)NNODES * DF;      // [E,128]
  float* cnt  = (float*)d_ws;                    // [N] edge counts

  gnn_zero_kernel<<<(NNODES * DF + 255) / 256, 256, 0, stream>>>(xout, cnt);
  gnn_edge_kernel<<<1024, 256, 0, stream>>>(x, ei, ea, We1, be1, We2, be2,
                                            eout, xout, cnt);
  gnn_node_kernel<<<(NNODES / 16 + 7) / 8, 256, 0, stream>>>(
      x, Wn1, bn1, Wn2, bn2, xout, cnt, xout);
}
